// CNN_79456894976579
// MI455X (gfx1250) — matrix operand order repaired and checked by reading, not compiled, not run on hardware
//
#include <hip/hip_runtime.h>

#define L     16384
#define LMASK (L - 1)
#define KT    4096
#define NB    32     // batch
#define NT    50     // templates
#define NTP   64     // padded template count

typedef __attribute__((ext_vector_type(16))) _Float16 v16h;
typedef __attribute__((ext_vector_type(8)))  _Float16 h8;
typedef __attribute__((ext_vector_type(8)))  float    v8f;

// Monotonic float<->uint mapping so unsigned atomicMax implements float max.
__device__ __forceinline__ unsigned enc_f32(float f) {
  unsigned u = __float_as_uint(f);
  return (u & 0x80000000u) ? ~u : (u | 0x80000000u);
}
__device__ __forceinline__ float dec_f32(unsigned k) {
  unsigned u = (k & 0x80000000u) ? (k & 0x7fffffffu) : ~k;
  return __uint_as_float(u);
}

__global__ void k_init(unsigned* __restrict__ z, int n) {
  int i = blockIdx.x * 256 + threadIdx.x;
  if (i < n) z[i] = 0u;  // encodes "below every real float"
}

__global__ void k_cvt_x(const float* __restrict__ xi, _Float16* __restrict__ x16, int n) {
  int i = blockIdx.x * 256 + threadIdx.x;
  if (i < n) x16[i] = (_Float16)xi[i];
}

// template [N,2,KT] f32  ->  h16 [2][NTP][KT] f16 (zero-padded n>=NT)
__global__ void k_cvt_h(const float* __restrict__ tpl, _Float16* __restrict__ h16) {
  int i = blockIdx.x * 256 + threadIdx.x;     // over 2*NTP*KT (exact multiple of 256)
  int ch = i / (NTP * KT);
  int r  = i % (NTP * KT);
  int n  = r / KT;
  int k  = r % KT;
  float v = (n < NT) ? tpl[((size_t)n * 2 + ch) * KT + k] : 0.f;
  h16[i] = (_Float16)v;
}

// Stage 1: d[b,ch][i] = sum_j S[ch][(j-i-1) mod L] * x[b,ch,j]
// GEMM per ch: A[i,j]=S[(j-i-1)%L] (circulant window from LDS), B[j,b]=x16, D[i,b].
// Writes d16 [B][2][L+64] f16 (64-elem halo = copy of first 64).
__global__ __launch_bounds__(256) void k_stage1(const float* __restrict__ S,
                                                const _Float16* __restrict__ x16,
                                                _Float16* __restrict__ d16) {
  __shared__ _Float16 sS[L + 64];
  const int ch  = blockIdx.y;
  const int tid = threadIdx.x;
  const float* Sg = S + (size_t)ch * L;
  for (int t = tid; t < L; t += 256) sS[t] = (_Float16)Sg[t];
  __syncthreads();
  if (tid < 64) sS[L + tid] = sS[tid];
  __syncthreads();

  const int wave = tid >> 5, lane = tid & 31;
  const int mrow = lane & 15;          // A row / B-col within tile
  const int khA  = (lane >> 4) << 4;   // A K base: 0 or 16
  const int khB  = (lane >> 4) << 4;   // B K base: 0 or 16
  const _Float16* xr0 = x16 + ((size_t)(mrow)      * 2 + ch) * L;  // b = mrow
  const _Float16* xr1 = x16 + ((size_t)(mrow + 16) * 2 + ch) * L;  // b = mrow+16

  for (int it = 0; it < 2; ++it) {
    const int i0 = (((int)blockIdx.x << 4) + (wave << 1) + it) << 4;  // i-tile base
    v8f c0 = {}, c1 = {};
    for (int j0 = 0; j0 < L; j0 += 32) {
      const int q = (j0 - i0 - 1 + 2 * L) & LMASK;   // scalar part of S index
      const int p = (q - mrow + L) & LMASK;          // per-lane base, halo covers +31
      v16h a;
#pragma unroll
      for (int e = 0; e < 8; ++e) a[e] = sS[p + khA + e];
#pragma unroll
      for (int e = 0; e < 8; ++e) a[8 + e] = sS[p + khA + 8 + e];
      v16h bf0, bf1;
      ((h8*)&bf0)[0] = *(const h8*)(xr0 + j0 + khB);
      ((h8*)&bf0)[1] = *(const h8*)(xr0 + j0 + khB + 8);
      ((h8*)&bf1)[0] = *(const h8*)(xr1 + j0 + khB);
      ((h8*)&bf1)[1] = *(const h8*)(xr1 + j0 + khB + 8);
      c0 = __builtin_amdgcn_wmma_f32_16x16x32_f16(false, a, false, bf0, (short)0, c0, false, false);
      c1 = __builtin_amdgcn_wmma_f32_16x16x32_f16(false, a, false, bf1, (short)0, c1, false, false);
    }
    // D layout: lanes 0-15 -> rows 0..7, lanes 16-31 -> rows 8..15; col = lane&15.
    const int roff = (lane >> 4) << 3;
    const int bcol = lane & 15;
    h8 hv0, hv1;
#pragma unroll
    for (int r = 0; r < 8; ++r) { hv0[r] = (_Float16)c0[r]; hv1[r] = (_Float16)c1[r]; }
    size_t o0 = ((size_t)(bcol * 2 + ch)) * (L + 64) + i0 + roff;
    size_t o1 = ((size_t)((bcol + 16) * 2 + ch)) * (L + 64) + i0 + roff;
    *(h8*)(d16 + o0) = hv0;
    *(h8*)(d16 + o1) = hv1;
    if (i0 < 64) { *(h8*)(d16 + o0 + L) = hv0; *(h8*)(d16 + o1 + L) = hv1; }
  }
}

// Stage 2: for each (b,ch): out[i,n] = sum_k d[(i+k-KT+1)%L] * h[n,k]; fused max over i.
// A = d window (LDS, halo), B = templates (global, L2-resident), 4 n-tiles share one A.
__global__ __launch_bounds__(256) void k_stage2(const _Float16* __restrict__ d16,
                                                const _Float16* __restrict__ h16,
                                                unsigned* __restrict__ zmax) {
  __shared__ _Float16 sd[L + 64];
  const int b   = blockIdx.y;
  const int ch  = blockIdx.z;
  const int tid = threadIdx.x;
  {
    const uint4* src = (const uint4*)(d16 + (size_t)(b * 2 + ch) * (L + 64));
    uint4* dst = (uint4*)sd;
    for (int t = tid; t < (L + 64) / 8; t += 256) dst[t] = src[t];
  }
  __syncthreads();

  const int wave = tid >> 5, lane = tid & 31;
  const int mrow = lane & 15;
  const int khA  = (lane >> 4) << 4;
  const int khB  = (lane >> 4) << 4;
  const _Float16* hb  = h16 + (size_t)ch * NTP * KT;
  const _Float16* hr0 = hb + (size_t)(0  + mrow) * KT + khB;
  const _Float16* hr1 = hb + (size_t)(16 + mrow) * KT + khB;
  const _Float16* hr2 = hb + (size_t)(32 + mrow) * KT + khB;
  const _Float16* hr3 = hb + (size_t)(48 + mrow) * KT + khB;

  float run0 = -3.4e38f, run1 = -3.4e38f, run2 = -3.4e38f, run3 = -3.4e38f;

  for (int it = 0; it < 8; ++it) {
    const int i0 = (((int)blockIdx.x << 6) + (it << 3) + wave) << 4;
    v8f c0 = {}, c1 = {}, c2 = {}, c3 = {};
    int s = (i0 + 1 - KT + L) & LMASK;     // scalar d-index base, wrapped once
    for (int k0 = 0; k0 < KT; k0 += 32) {
      const int p = s + mrow + khA;        // reads up to p+15 <= L+45 < L+64 (halo)
      v16h a;
#pragma unroll
      for (int e = 0; e < 8; ++e) a[e] = sd[p + e];
#pragma unroll
      for (int e = 0; e < 8; ++e) a[8 + e] = sd[p + 8 + e];
      v16h bf;
      ((h8*)&bf)[0] = *(const h8*)(hr0 + k0);
      ((h8*)&bf)[1] = *(const h8*)(hr0 + k0 + 8);
      c0 = __builtin_amdgcn_wmma_f32_16x16x32_f16(false, a, false, bf, (short)0, c0, false, false);
      ((h8*)&bf)[0] = *(const h8*)(hr1 + k0);
      ((h8*)&bf)[1] = *(const h8*)(hr1 + k0 + 8);
      c1 = __builtin_amdgcn_wmma_f32_16x16x32_f16(false, a, false, bf, (short)0, c1, false, false);
      ((h8*)&bf)[0] = *(const h8*)(hr2 + k0);
      ((h8*)&bf)[1] = *(const h8*)(hr2 + k0 + 8);
      c2 = __builtin_amdgcn_wmma_f32_16x16x32_f16(false, a, false, bf, (short)0, c2, false, false);
      ((h8*)&bf)[0] = *(const h8*)(hr3 + k0);
      ((h8*)&bf)[1] = *(const h8*)(hr3 + k0 + 8);
      c3 = __builtin_amdgcn_wmma_f32_16x16x32_f16(false, a, false, bf, (short)0, c3, false, false);
      s += 32; if (s >= L) s -= L;
    }
#pragma unroll
    for (int r = 0; r < 8; ++r) {
      run0 = fmaxf(run0, c0[r]); run1 = fmaxf(run1, c1[r]);
      run2 = fmaxf(run2, c2[r]); run3 = fmaxf(run3, c3[r]);
    }
  }
  // rows 0..7 live in lanes 0-15, rows 8..15 in lanes 16-31 (same n col): combine.
  run0 = fmaxf(run0, __shfl_xor(run0, 16));
  run1 = fmaxf(run1, __shfl_xor(run1, 16));
  run2 = fmaxf(run2, __shfl_xor(run2, 16));
  run3 = fmaxf(run3, __shfl_xor(run3, 16));
  if (lane < 16) {
    unsigned* zr = zmax + (((size_t)b * 2 + ch) << 6);
    atomicMax(zr + 0  + lane, enc_f32(run0));
    atomicMax(zr + 16 + lane, enc_f32(run1));
    atomicMax(zr + 32 + lane, enc_f32(run2));
    atomicMax(zr + 48 + lane, enc_f32(run3));
  }
}

// Tiny head: z = max/hh -> conv(3)+sigmoid+pool3 -> conv(3)+sigmoid+pool3 -> FC relu -> FC
__global__ __launch_bounds__(64) void k_head(const unsigned* __restrict__ zenc,
    const float* __restrict__ hh,
    const float* __restrict__ W1, const float* __restrict__ B1,
    const float* __restrict__ W2, const float* __restrict__ B2,
    const float* __restrict__ W3, const float* __restrict__ B3,
    const float* __restrict__ W4, const float* __restrict__ B4,
    float* __restrict__ out) {
  const int b = blockIdx.x, tid = threadIdx.x;
  __shared__ float z[2][NT];
  __shared__ float c1s[16][48];
  __shared__ float p1s[16][16];
  __shared__ float c2s[32][14];
  __shared__ float fl[128];
  __shared__ float f3[32];

  for (int i = tid; i < 2 * NT; i += 64) {
    int chn = i / NT, n = i % NT;
    float v = dec_f32(zenc[(((size_t)b * 2 + chn) << 6) + n]);
    z[chn][n] = v / hh[n * 2 + chn];
  }
  __syncthreads();
  for (int i = tid; i < 16 * 48; i += 64) {
    int o = i / 48, p = i % 48;
    float s = B1[o];
#pragma unroll
    for (int c = 0; c < 2; ++c)
#pragma unroll
      for (int t = 0; t < 3; ++t) s += W1[(o * 2 + c) * 3 + t] * z[c][p + t];
    c1s[o][p] = 1.f / (1.f + expf(-s));
  }
  __syncthreads();
  for (int i = tid; i < 16 * 16; i += 64) {
    int o = i / 16, q = i % 16;
    p1s[o][q] = fmaxf(fmaxf(c1s[o][3 * q], c1s[o][3 * q + 1]), c1s[o][3 * q + 2]);
  }
  __syncthreads();
  for (int i = tid; i < 32 * 14; i += 64) {
    int o = i / 14, p = i % 14;
    float s = B2[o];
    for (int c = 0; c < 16; ++c)
#pragma unroll
      for (int t = 0; t < 3; ++t) s += W2[(o * 16 + c) * 3 + t] * p1s[c][p + t];
    c2s[o][p] = 1.f / (1.f + expf(-s));
  }
  __syncthreads();
  for (int i = tid; i < 128; i += 64) {
    int o = i / 4, q = i % 4;
    fl[i] = fmaxf(fmaxf(c2s[o][3 * q], c2s[o][3 * q + 1]), c2s[o][3 * q + 2]);
  }
  __syncthreads();
  if (tid < 32) {
    float s = B3[tid];
    for (int v = 0; v < 128; ++v) s += W3[tid * 128 + v] * fl[v];
    f3[tid] = fmaxf(s, 0.f);
  }
  __syncthreads();
  if (tid < 2) {
    float s = B4[tid];
    for (int u = 0; u < 32; ++u) s += W4[tid * 32 + u] * f3[u];
    out[b * 2 + tid] = s;
  }
}

extern "C" void kernel_launch(void* const* d_in, const int* in_sizes, int n_in,
                              void* d_out, int out_size, void* d_ws, size_t ws_size,
                              hipStream_t stream) {
  const float* xi  = (const float*)d_in[0];
  const float* S   = (const float*)d_in[1];
  const float* tpl = (const float*)d_in[2];
  const float* hh  = (const float*)d_in[3];
  const float* W1  = (const float*)d_in[4];
  const float* B1  = (const float*)d_in[5];
  const float* W2  = (const float*)d_in[6];
  const float* B2  = (const float*)d_in[7];
  const float* W3  = (const float*)d_in[8];
  const float* B3  = (const float*)d_in[9];
  const float* W4  = (const float*)d_in[10];
  const float* B4  = (const float*)d_in[11];
  float* out = (float*)d_out;

  char* ws = (char*)d_ws;
  _Float16* x16 = (_Float16*)(ws);                               // NB*2*L*2      = 2 MiB
  _Float16* h16 = (_Float16*)(ws + (size_t)2097152);             // 2*NTP*KT*2    = 1 MiB
  _Float16* d16 = (_Float16*)(ws + (size_t)3145728);             // NB*2*(L+64)*2 = 2,105,344 B
  unsigned* zme = (unsigned*)(ws + (size_t)5251072);             // NB*2*NTP*4    = 16 KiB

  k_init <<<dim3((NB * 2 * NTP + 255) / 256), 256, 0, stream>>>(zme, NB * 2 * NTP);
  k_cvt_x<<<dim3((NB * 2 * L) / 256),          256, 0, stream>>>(xi, x16, NB * 2 * L);
  k_cvt_h<<<dim3((2 * NTP * KT) / 256),        256, 0, stream>>>(tpl, h16);
  k_stage1<<<dim3(64, 2),    256, 0, stream>>>(S, x16, d16);
  k_stage2<<<dim3(16, NB, 2), 256, 0, stream>>>(d16, h16, zme);
  k_head <<<dim3(NB), 64, 0, stream>>>(zme, hh, W1, B1, W2, B2, W3, B3, W4, B4, out);
}
